// HeteroGNN_38835094291148
// MI455X (gfx1250) — matrix operand order repaired and checked by reading, not compiled, not run on hardware
//
#include <hip/hip_runtime.h>
#include <hip/hip_bf16.h>

// ---------------------------------------------------------------------------
// HeteroGNN forward for MI455X (gfx1250, wave32, WMMA).
//
// ~123 GFLOP dense GEMM (dominated by 50000x1900x256 gene encoder) on
// v_wmma_f32_16x16x32_bf16 (fp32 loads, in-register bf16 convert, f32 acc);
// ~5 GB of edge gather/scatter bounded by the 23.3 TB/s HBM roofline.
//
// GEMM K-loop: unguarded b128 (float4) loads in the main body, one guarded
// tail iteration (only K=1900 has a tail). Each wave computes a 16x64 C strip
// (1 A fragment reused across 4 WMMAs per K-step).
// ---------------------------------------------------------------------------

#define H_DIM 256
#define NG 50000
#define NM 30000
#define EPP 500000
#define EMM 300000
#define EGM 600000

typedef __attribute__((ext_vector_type(16))) __bf16 v16bf;
typedef __attribute__((ext_vector_type(8)))  float  v8f;
typedef __attribute__((ext_vector_type(4)))  float  f32x4;

// A fragment (16-bit A 16x32; contiguous k order chosen here, same as B): lane -> M = lane&15, half = lane>>4;
// element e -> K = half*16 + e. Per lane: 16 contiguous
// floats -> four float4 loads.
__device__ __forceinline__ v16bf load_a_frag(const float* __restrict__ arow,
                                             int k0, int half) {
  const f32x4* p0 = (const f32x4*)(arow + k0 + (half << 4));
  const f32x4* p1 = (const f32x4*)(arow + k0 + 8 + (half << 4));
  f32x4 a0 = p0[0], a1 = p0[1], a2 = p1[0], a3 = p1[1];
  v16bf f;
#pragma unroll
  for (int i = 0; i < 4; ++i) {
    f[i]      = (__bf16)a0[i];
    f[4 + i]  = (__bf16)a1[i];
    f[8 + i]  = (__bf16)a2[i];
    f[12 + i] = (__bf16)a3[i];
  }
  return f;
}

// B fragment (16-bit B 32x16): lane -> N = lane&15; element e -> K = half*16+e.
// 16 contiguous floats per lane -> four float4 loads from the transposed weight.
__device__ __forceinline__ v16bf load_b_frag(const float* __restrict__ brow,
                                             int k0, int half) {
  const f32x4* p = (const f32x4*)(brow + k0 + (half << 4));
  f32x4 b0 = p[0], b1 = p[1], b2 = p[2], b3 = p[3];
  v16bf f;
#pragma unroll
  for (int i = 0; i < 4; ++i) {
    f[i]      = (__bf16)b0[i];
    f[4 + i]  = (__bf16)b1[i];
    f[8 + i]  = (__bf16)b2[i];
    f[12 + i] = (__bf16)b3[i];
  }
  return f;
}

// Guarded tail fragments (taken at most once, only when K % 32 != 0).
__device__ __forceinline__ v16bf load_a_tail(const float* __restrict__ arow,
                                             int k0, int half, int K) {
  v16bf f;
#pragma unroll
  for (int e = 0; e < 16; ++e) {
    int ka = k0 + (half << 4) + e;
    f[e] = (ka < K) ? (__bf16)arow[ka] : (__bf16)0.0f;
  }
  return f;
}
__device__ __forceinline__ v16bf load_b_tail(const float* __restrict__ brow,
                                             int k0, int half, int K) {
  v16bf f;
#pragma unroll
  for (int e = 0; e < 16; ++e) {
    int kb = k0 + (half << 4) + e;
    f[e] = (kb < K) ? (__bf16)brow[kb] : (__bf16)0.0f;
  }
  return f;
}

// ---------------------------------------------------------------------------
// WMMA GEMM: C[M,N] = A[M,K] (f32 row-major) x B[K,N], B given transposed as
// Bt[N,K]. Block = 128 threads = 4 waves; wave w computes rows
// [blockIdx.x*64 + w*16, +16) x cols [blockIdx.y*64, +64) -> 4 accumulators.
// Requires M % 16 == 0 and N % 64 == 0 (true for all call sites: M in
// {50000, 30000}, N = 256). C/D layout: element r -> M = half*8 + r,
// N = lane&15.
// ---------------------------------------------------------------------------
__global__ void gemm_bf16_wmma(const float* __restrict__ A,
                               const float* __restrict__ Bt,
                               float* __restrict__ C,
                               int M, int N, int K) {
  const int lane = threadIdx.x & 31;
  const int wave = threadIdx.x >> 5;   // 0..3
  const int half = lane >> 4;
  const int mr   = lane & 15;

  const int m0 = (blockIdx.x * 4 + wave) * 16;  // wave-uniform
  const int n0 = blockIdx.y * 64;
  if (m0 >= M) return;                          // uniform exit: EXEC stays full

  const float* __restrict__ arow = A + (size_t)(m0 + mr) * K;
  const float* __restrict__ br0  = Bt + (size_t)(n0 + mr) * K;
  const size_t bstep = (size_t)16 * K;          // next 16-col tile of Bt

  v8f acc0 = {}, acc1 = {}, acc2 = {}, acc3 = {};
  const int Kmain = K & ~31;
  int k0 = 0;
  for (; k0 < Kmain; k0 += 32) {
    __builtin_prefetch(arow + k0 + 128, 0, 1);  // global_prefetch_b8
    v16bf af = load_a_frag(arow, k0, half);
    v16bf b0 = load_b_frag(br0,             k0, half);
    v16bf b1 = load_b_frag(br0 + bstep,     k0, half);
    v16bf b2 = load_b_frag(br0 + 2 * bstep, k0, half);
    v16bf b3 = load_b_frag(br0 + 3 * bstep, k0, half);
    acc0 = __builtin_amdgcn_wmma_f32_16x16x32_bf16(false, af, false, b0, (short)0, acc0, false, false);
    acc1 = __builtin_amdgcn_wmma_f32_16x16x32_bf16(false, af, false, b1, (short)0, acc1, false, false);
    acc2 = __builtin_amdgcn_wmma_f32_16x16x32_bf16(false, af, false, b2, (short)0, acc2, false, false);
    acc3 = __builtin_amdgcn_wmma_f32_16x16x32_bf16(false, af, false, b3, (short)0, acc3, false, false);
  }
  if (k0 < K) {                                 // single guarded tail step
    v16bf af = load_a_tail(arow, k0, half, K);
    v16bf b0 = load_b_tail(br0,             k0, half, K);
    v16bf b1 = load_b_tail(br0 + bstep,     k0, half, K);
    v16bf b2 = load_b_tail(br0 + 2 * bstep, k0, half, K);
    v16bf b3 = load_b_tail(br0 + 3 * bstep, k0, half, K);
    acc0 = __builtin_amdgcn_wmma_f32_16x16x32_bf16(false, af, false, b0, (short)0, acc0, false, false);
    acc1 = __builtin_amdgcn_wmma_f32_16x16x32_bf16(false, af, false, b1, (short)0, acc1, false, false);
    acc2 = __builtin_amdgcn_wmma_f32_16x16x32_bf16(false, af, false, b2, (short)0, acc2, false, false);
    acc3 = __builtin_amdgcn_wmma_f32_16x16x32_bf16(false, af, false, b3, (short)0, acc3, false, false);
  }

  float* crow = C + (size_t)(m0 + (half << 3)) * N + n0 + mr;
#pragma unroll
  for (int r = 0; r < 8; ++r) {
    crow[(size_t)r * N + 0]  = acc0[r];
    crow[(size_t)r * N + 16] = acc1[r];
    crow[(size_t)r * N + 32] = acc2[r];
    crow[(size_t)r * N + 48] = acc3[r];
  }
}

// ---------------------------------------------------------------------------
// Small utility kernels
// ---------------------------------------------------------------------------
__global__ void transpose_f32(const float* __restrict__ W, float* __restrict__ Wt,
                              int rows, int cols) {
  int idx = blockIdx.x * blockDim.x + threadIdx.x;
  if (idx >= rows * cols) return;
  int r = idx / cols, c = idx - r * cols;
  Wt[(size_t)c * rows + r] = W[idx];
}

// Biased batch statistics per column (BatchNorm1d training mode).
__global__ void col_stats(const float* __restrict__ X, int rows, int cols,
                          float* __restrict__ mean, float* __restrict__ rstd) {
  const int j = blockIdx.x;
  float s = 0.f, ss = 0.f;
  for (int i = threadIdx.x; i < rows; i += blockDim.x) {
    float v = X[(size_t)i * cols + j];
    s += v; ss += v * v;
  }
  __shared__ float sh0[256], sh1[256];
  sh0[threadIdx.x] = s; sh1[threadIdx.x] = ss;
  __syncthreads();
  for (int off = 128; off > 0; off >>= 1) {
    if (threadIdx.x < off) {
      sh0[threadIdx.x] += sh0[threadIdx.x + off];
      sh1[threadIdx.x] += sh1[threadIdx.x + off];
    }
    __syncthreads();
  }
  if (threadIdx.x == 0) {
    float mu  = sh0[0] / (float)rows;
    float var = sh1[0] / (float)rows - mu * mu;
    mean[j] = mu;
    rstd[j] = rsqrtf(var + 1e-5f);
  }
}

// relu(g*(x-mu)*rstd + beta). (Linear bias cancels inside BN: mean shift.)
__global__ void bn_relu(float* __restrict__ X, const float* __restrict__ mean,
                        const float* __restrict__ rstd, const float* __restrict__ g,
                        const float* __restrict__ beta, int rows, int cols) {
  size_t idx = (size_t)blockIdx.x * blockDim.x + threadIdx.x;
  if (idx >= (size_t)rows * cols) return;
  int j = (int)(idx % cols);
  float v = g[j] * (X[idx] - mean[j]) * rstd[j] + beta[j];
  X[idx] = v > 0.f ? v : 0.f;
}

__global__ void bias_relu(float* __restrict__ X, const float* __restrict__ b,
                          int rows, int cols) {
  size_t idx = (size_t)blockIdx.x * blockDim.x + threadIdx.x;
  if (idx >= (size_t)rows * cols) return;
  float v = X[idx] + b[idx % cols];
  X[idx] = v > 0.f ? v : 0.f;
}

// out[i] = dot(X[i,:], a)  -- one wave32 per row.
__global__ void row_dot(const float* __restrict__ X, const float* __restrict__ a,
                        float* __restrict__ out, int n, int h) {
  int gid  = blockIdx.x * blockDim.x + threadIdx.x;
  int wid  = gid >> 5;
  int lane = gid & 31;
  if (wid >= n) return;
  const float* xr = X + (size_t)wid * h;
  float s = 0.f;
  for (int j = lane; j < h; j += 32) s += xr[j] * a[j];
  for (int off = 16; off > 0; off >>= 1) s += __shfl_down(s, off, 32);
  if (lane == 0) out[wid] = s;
}

// Order-preserving float<->uint encoding so atomicMax(u32) == float max.
__device__ __forceinline__ unsigned fenc(float f) {
  unsigned u = __float_as_uint(f);
  return (u & 0x80000000u) ? ~u : (u | 0x80000000u);
}
__device__ __forceinline__ float fdec(unsigned u) {
  u = (u & 0x80000000u) ? (u & 0x7fffffffu) : ~u;
  return __uint_as_float(u);
}
__device__ __forceinline__ float lrelu02(float a) { return a > 0.f ? a : 0.2f * a; }

// ---- GAT edge passes (recompute edge logit each pass; cheaper than storage)
__global__ void gat_edge_max(const int* __restrict__ row, const int* __restrict__ col,
                             const float* __restrict__ ssrc, const float* __restrict__ sdst,
                             unsigned* __restrict__ segmax, int E) {
  int e = blockIdx.x * blockDim.x + threadIdx.x;
  if (e >= E) return;
  float a = lrelu02(ssrc[row[e]] + sdst[col[e]]);
  atomicMax(&segmax[col[e]], fenc(a));
}

__global__ void gat_edge_expsum(const int* __restrict__ row, const int* __restrict__ col,
                                const float* __restrict__ ssrc, const float* __restrict__ sdst,
                                const unsigned* __restrict__ segmax,
                                float* __restrict__ segsum, int E) {
  int e = blockIdx.x * blockDim.x + threadIdx.x;
  if (e >= E) return;
  int c = col[e];
  float a = lrelu02(ssrc[row[e]] + sdst[c]);
  atomicAdd(&segsum[c], expf(a - fdec(segmax[c])));
}

// Wave-per-edge: out[col] += alpha * hs[row]  (256 floats, 8 atomics/lane)
__global__ void gat_edge_scatter(const int* __restrict__ row, const int* __restrict__ col,
                                 const float* __restrict__ ssrc, const float* __restrict__ sdst,
                                 const unsigned* __restrict__ segmax,
                                 const float* __restrict__ segsum,
                                 const float* __restrict__ hs, float* __restrict__ out,
                                 int E, int h) {
  int gid  = blockIdx.x * blockDim.x + threadIdx.x;
  int e    = gid >> 5;
  int lane = gid & 31;
  if (e >= E) return;
  int r = row[e], c = col[e];
  float a = lrelu02(ssrc[r] + sdst[c]);
  float alpha = expf(a - fdec(segmax[c])) / segsum[c];
  const float* src = hs + (size_t)r * h;
  float* dst = out + (size_t)c * h;
  for (int j = lane; j < h; j += 32) atomicAdd(&dst[j], alpha * src[j]);
}

// ---- GCN passes
__global__ void edge_deg(const int* __restrict__ col, float* __restrict__ deg, int E) {
  int e = blockIdx.x * blockDim.x + threadIdx.x;
  if (e >= E) return;
  atomicAdd(&deg[col[e]], 1.0f);
}
__global__ void make_dinv(float* __restrict__ deg, int n) {
  int i = blockIdx.x * blockDim.x + threadIdx.x;
  if (i >= n) return;
  float d = deg[i];
  deg[i] = d > 0.f ? rsqrtf(d) : 0.f;
}
__global__ void gcn_edge_scatter(const int* __restrict__ row, const int* __restrict__ col,
                                 const float* __restrict__ dinv,
                                 const float* __restrict__ hsrc, float* __restrict__ out,
                                 int E, int h) {
  int gid  = blockIdx.x * blockDim.x + threadIdx.x;
  int e    = gid >> 5;
  int lane = gid & 31;
  if (e >= E) return;
  int r = row[e], c = col[e];
  float norm = dinv[r] * dinv[c];
  const float* src = hsrc + (size_t)r * h;
  float* dst = out + (size_t)c * h;
  for (int j = lane; j < h; j += 32) atomicAdd(&dst[j], norm * src[j]);
}

// ---------------------------------------------------------------------------
// Host orchestration
// ---------------------------------------------------------------------------
extern "C" void kernel_launch(void* const* d_in, const int* in_sizes, int n_in,
                              void* d_out, int out_size, void* d_ws, size_t ws_size,
                              hipStream_t stream) {
  const int H = H_DIM;
  // d_in layout (setup_inputs dict insertion order, params flattened in
  // nested insertion order):
  //  0 x_gene [NG,1900]   1 x_mesh [NM,768]
  //  2..5  enc.gene {W[1900,256], b, g, beta}
  //  6..9  enc.mesh {W[768,256],  b, g, beta}
  //  10..14 gat0.gm {Ws,Wd,as,ad,b}   15..19 gat0.mg {...}
  //  20..24 gat1.gm                   25..29 gat1.mg
  //  30,31 gcn0.gg {W,b}  32,33 gcn0.mm  34,35 gcn1.gg  36,37 gcn1.mm
  //  38 ei_ppi [2,EPP]  39 ei_mm [2,EMM]  40 ei_gm [2,EGM]  41 ei_mg [2,EGM]
  const float* x_gene = (const float*)d_in[0];
  const float* x_mesh = (const float*)d_in[1];
  const int* ei_ppi = (const int*)d_in[38];
  const int* ei_mm  = (const int*)d_in[39];
  const int* ei_gm  = (const int*)d_in[40];
  const int* ei_mg  = (const int*)d_in[41];

  const size_t NGH = (size_t)NG * H;
  const size_t NMH = (size_t)NM * H;

  float* w = (float*)d_ws;
  size_t off = 0;
  auto alloc = [&](size_t n) { float* p = w + off; off += n; return p; };
  float* gA     = alloc(NGH);
  float* gB     = alloc(NGH);
  float* mA     = alloc(NMH);
  float* mB     = alloc(NMH);
  float* hsBuf  = alloc(NGH);         // src projection (max NG rows)
  float* hdBuf  = alloc(NGH);         // dst projection (max NG rows)
  float* Wt     = alloc((size_t)1900 * H);   // largest transposed weight
  float* ssrc   = alloc(NG);
  float* sdst   = alloc(NG);
  unsigned* segmax = (unsigned*)alloc(NG);
  float* segsum = alloc(NG);
  float* deg    = alloc(NG);
  float* meanB  = alloc(H);
  float* rstdB  = alloc(H);
  (void)ws_size; (void)n_in; (void)in_sizes; (void)out_size;

  auto gemm = [&](const float* A, const float* Wsrc, int M, int N, int K, float* C) {
    int tn = K * N;
    transpose_f32<<<(tn + 255) / 256, 256, 0, stream>>>(Wsrc, Wt, K, N);
    dim3 grid((M / 16 + 3) / 4, N / 64);
    gemm_bf16_wmma<<<grid, 128, 0, stream>>>(A, Wt, C, M, N, K);
  };
  auto ew_blocks = [](size_t n) { return (unsigned)((n + 255) / 256); };

  // ---------------- Encoders: GEMM -> BN(train stats) -> ReLU -------------
  gemm(x_gene, (const float*)d_in[2], NG, H, 1900, gA);
  col_stats<<<H, 256, 0, stream>>>(gA, NG, H, meanB, rstdB);
  bn_relu<<<ew_blocks(NGH), 256, 0, stream>>>(gA, meanB, rstdB,
      (const float*)d_in[4], (const float*)d_in[5], NG, H);

  gemm(x_mesh, (const float*)d_in[6], NM, H, 768, mA);
  col_stats<<<H, 256, 0, stream>>>(mA, NM, H, meanB, rstdB);
  bn_relu<<<ew_blocks(NMH), 256, 0, stream>>>(mA, meanB, rstdB,
      (const float*)d_in[8], (const float*)d_in[9], NM, H);

  float* gcur = gA; float* gnxt = gB;
  float* mcur = mA; float* mnxt = mB;

  auto gat_rel = [&](const float* xs, int ns, const float* xd, int nd,
                     int pbase, const int* erow, const int* ecol, int E, float* out) {
    const float* Ws = (const float*)d_in[pbase + 0];
    const float* Wd = (const float*)d_in[pbase + 1];
    const float* av = (const float*)d_in[pbase + 2];
    const float* ad = (const float*)d_in[pbase + 3];
    const float* b  = (const float*)d_in[pbase + 4];
    gemm(xs, Ws, ns, H, H, hsBuf);
    gemm(xd, Wd, nd, H, H, hdBuf);
    row_dot<<<(ns * 32 + 255) / 256, 256, 0, stream>>>(hsBuf, av, ssrc, ns, H);
    row_dot<<<(nd * 32 + 255) / 256, 256, 0, stream>>>(hdBuf, ad, sdst, nd, H);
    hipMemsetAsync(segmax, 0, (size_t)nd * 4, stream);  // fenc-space -inf
    hipMemsetAsync(segsum, 0, (size_t)nd * 4, stream);
    gat_edge_max<<<(E + 255) / 256, 256, 0, stream>>>(erow, ecol, ssrc, sdst, segmax, E);
    gat_edge_expsum<<<(E + 255) / 256, 256, 0, stream>>>(erow, ecol, ssrc, sdst, segmax, segsum, E);
    hipMemsetAsync(out, 0, (size_t)nd * H * 4, stream);
    gat_edge_scatter<<<((size_t)E * 32 + 255) / 256, 256, 0, stream>>>(
        erow, ecol, ssrc, sdst, segmax, segsum, hsBuf, out, E, H);
    bias_relu<<<ew_blocks((size_t)nd * H), 256, 0, stream>>>(out, b, nd, H);
  };

  auto gcn_rel = [&](const float* x, int n, int pbase, const int* erow,
                     const int* ecol, int E, float* out) {
    const float* W = (const float*)d_in[pbase + 0];
    const float* b = (const float*)d_in[pbase + 1];
    gemm(x, W, n, H, H, hsBuf);
    hipMemsetAsync(deg, 0, (size_t)n * 4, stream);
    edge_deg<<<(E + 255) / 256, 256, 0, stream>>>(ecol, deg, E);
    make_dinv<<<(n + 255) / 256, 256, 0, stream>>>(deg, n);
    hipMemsetAsync(out, 0, (size_t)n * H * 4, stream);
    gcn_edge_scatter<<<((size_t)E * 32 + 255) / 256, 256, 0, stream>>>(
        erow, ecol, deg, hsBuf, out, E, H);
    bias_relu<<<ew_blocks((size_t)n * H), 256, 0, stream>>>(out, b, n, H);
  };

  // ---------------- 2x GAT hetero layers ----------------------------------
  for (int l = 0; l < 2; ++l) {
    int base = 10 + l * 10;
    // gm: src=gene, dst=mesh  -> m2
    gat_rel(gcur, NG, mcur, NM, base + 0, ei_gm, ei_gm + EGM, EGM, mnxt);
    // mg: src=mesh, dst=gene  -> g2
    gat_rel(mcur, NM, gcur, NG, base + 5, ei_mg, ei_mg + EGM, EGM, gnxt);
    float* t;
    t = gcur; gcur = gnxt; gnxt = t;
    t = mcur; mcur = mnxt; mnxt = t;
  }

  // ---------------- 2x GCN hetero layers ----------------------------------
  for (int l = 0; l < 2; ++l) {
    int base = 30 + l * 4;
    gcn_rel(gcur, NG, base + 0, ei_ppi, ei_ppi + EPP, EPP, gnxt);
    gcn_rel(mcur, NM, base + 2, ei_mm, ei_mm + EMM, EMM, mnxt);
    float* t;
    t = gcur; gcur = gnxt; gnxt = t;
    t = mcur; mcur = mnxt; mnxt = t;
  }

  // ---------------- Output: concat(g, m) ----------------------------------
  hipMemcpyAsync(d_out, gcur, NGH * sizeof(float), hipMemcpyDeviceToDevice, stream);
  hipMemcpyAsync((float*)d_out + NGH, mcur, NMH * sizeof(float),
                 hipMemcpyDeviceToDevice, stream);
}
